// TransNetExplicit2_41077067219034
// MI455X (gfx1250) — compile-verified, matrix operand order checked by reading, not run on hardware
//
#include <hip/hip_runtime.h>
#include <math.h>

// ---------------- problem constants ----------------
#define UNITS      128
#define INPUT_DIM  784
#define OUTPUT_DIM 10
#define STEPS      20
#define NBLOCKS    4
#define DT_C       0.05f           // 1/20
#define CC_C       5.0f            // DT/EPS
#define SCALE_C    (1.0f/6.0f)     // 1/(1+c)

#define TILE_ROWS  256             // batch rows per workgroup (8 waves x 32)
#define NT         2               // batch 16-row N-tiles per wave
#define NTILE      8               // 128 units / 16
#define NKC        4               // 128 / 32 K-chunks

typedef __bf16 bf16_t;
typedef __attribute__((ext_vector_type(16))) __bf16 v16bf;
typedef __attribute__((ext_vector_type(8)))  __bf16 v8bf;
typedef __attribute__((ext_vector_type(8)))  float  v8f;

union BF16x16 { v16bf v; v8bf h[2]; };

// A-operand fragment from two contiguous 16-byte LDS runs
// (16-bit layout: element e <-> K = (e<8?0:16) + 8*hs + (e&7))
__device__ __forceinline__ v16bf lds_frag(const bf16_t* p0) {
  BF16x16 r;
  r.h[0] = *(const v8bf*)p0;
  r.h[1] = *(const v8bf*)(p0 + 16);
  return r.v;
}

// B-operand fragment for one K-chunk built from two C-layout f32 tiles
// of the same lane (registers 0..7 of each) -- in-register bf16 convert.
__device__ __forceinline__ v16bf state_frag(const v8f a, const v8f b) {
  v16bf f;
#pragma unroll
  for (int e = 0; e < 8; ++e) { f[e] = (bf16_t)a[e]; f[e + 8] = (bf16_t)b[e]; }
  return f;
}

__device__ __forceinline__ v8f load_bias8(const float* b) {
  float4 lo = *(const float4*)(b);
  float4 hi = *(const float4*)(b + 4);
  v8f a; a[0]=lo.x; a[1]=lo.y; a[2]=lo.z; a[3]=lo.w; a[4]=hi.x; a[5]=hi.y; a[6]=hi.z; a[7]=hi.w;
  return a;
}

// B-frag of x^T from global (f32 -> bf16), zero-padded past INPUT_DIM
__device__ __forceinline__ v16bf xfrag(const float* xr, int k0) {
  BF16x16 bf;
#pragma unroll
  for (int half = 0; half < 2; ++half) {
    int kk = k0 + 16 * half;
    v8bf hv;
    if (kk < INPUT_DIM) {
      float4 f0 = *(const float4*)(xr + kk);
      float4 f1 = *(const float4*)(xr + kk + 4);
      hv[0]=(bf16_t)f0.x; hv[1]=(bf16_t)f0.y; hv[2]=(bf16_t)f0.z; hv[3]=(bf16_t)f0.w;
      hv[4]=(bf16_t)f1.x; hv[5]=(bf16_t)f1.y; hv[6]=(bf16_t)f1.z; hv[7]=(bf16_t)f1.w;
    } else {
#pragma unroll
      for (int e = 0; e < 8; ++e) hv[e] = (bf16_t)0.0f;
    }
    bf.h[half] = hv;
  }
  return bf.v;
}

__device__ __forceinline__ float fast_tanh(float x) {
  float e = __expf(2.0f * x);                       // saturates correctly
  return 1.0f - 2.0f * __builtin_amdgcn_rcpf(e + 1.0f);
}

__global__ __launch_bounds__(256, 1)
void transnet_kernel(const float* __restrict__ x,
                     const float* __restrict__ Wi, const float* __restrict__ bi,
                     const float* __restrict__ W1, const float* __restrict__ b1,
                     const float* __restrict__ W2, const float* __restrict__ b2,
                     const float* __restrict__ W3, const float* __restrict__ b3,
                     const float* __restrict__ W4, const float* __restrict__ b4,
                     const float* __restrict__ Wo, const float* __restrict__ bo,
                     float* __restrict__ out)
{
  // ---- LDS carve (~133 KB) ----
  __shared__ __align__(16) unsigned char smem[136320];
  bf16_t* sWA = (bf16_t*)(smem);            // [m][k] = W[k][m]  (A-frags, phase A); also Wi chunk
  bf16_t* sWB = (bf16_t*)(smem + 32768);    // [m][k] = W[m][k]  (A-frags, phase B)
  float*  sBo = (float*)(smem + 131072);    // 16 (10 used)
  float*  sWo = (float*)(smem + 131136);    // 10*128
  float*  sOutF = (float*)(smem);           // f32[256][128], reused after recurrence

  const int tid  = threadIdx.x;
  const int wave = tid >> 5;
  const int lane = tid & 31;
  const int lr   = lane & 15;   // A-frag row (m) / B,C,D column within tile
  const int hs   = lane >> 4;   // half select
  const int rowBase = blockIdx.x * TILE_ROWS + wave * 32;

  // ---- stage head constants ----
  for (int i = tid; i < OUTPUT_DIM * UNITS; i += 256) sWo[i] = Wo[i];
  if (tid < OUTPUT_DIM) sBo[tid] = bo[tid];

  // =======================================================================
  // Input layer (transposed): z^T = Wi @ x^T, M = units, N = 2 batch tiles
  // =======================================================================
  v8f bias_v[NTILE];
#pragma unroll
  for (int t = 0; t < NTILE; ++t) bias_v[t] = load_bias8(bi + t * 16 + 8 * hs);

  v8f uacc[NT][NTILE];
#pragma unroll
  for (int n = 0; n < NT; ++n)
#pragma unroll
    for (int t = 0; t < NTILE; ++t) uacc[n][t] = bias_v[t];

  const int NKC_IN = (INPUT_DIM + 31) / 32;                        // 25
  const float* xr0 = x + (size_t)(rowBase + lr) * INPUT_DIM;
  const float* xr1 = x + (size_t)(rowBase + 16 + lr) * INPUT_DIM;
  for (int kc = 0; kc < NKC_IN; ++kc) {
    __syncthreads();   // protect sWA overwrite
    for (int idx = tid; idx < UNITS * 32; idx += 256) {            // sWA[m][kk]=Wi[m][kc*32+kk]
      int m = idx >> 5, kk = idx & 31, k = kc * 32 + kk;
      sWA[m * 32 + kk] = (k < INPUT_DIM) ? (bf16_t)Wi[m * INPUT_DIM + k] : (bf16_t)0.0f;
    }
    __syncthreads();

    int k0 = kc * 32 + 8 * hs;
    v16bf bfx0 = xfrag(xr0, k0);
    v16bf bfx1 = xfrag(xr1, k0);
    if (kc + 1 < NKC_IN) { __builtin_prefetch(xr0 + (kc + 1) * 32, 0, 0);
                           __builtin_prefetch(xr1 + (kc + 1) * 32, 0, 0); }

    v16bf aw[NTILE];
#pragma unroll
    for (int t = 0; t < NTILE; ++t) aw[t] = lds_frag(sWA + (t * 16 + lr) * 32 + 8 * hs);
#pragma unroll
    for (int t = 0; t < NTILE; ++t) {
      uacc[0][t] = __builtin_amdgcn_wmma_f32_16x16x32_bf16(
          false, aw[t], false, bfx0, (short)0, uacc[0][t], false, false);
      uacc[1][t] = __builtin_amdgcn_wmma_f32_16x16x32_bf16(
          false, aw[t], false, bfx1, (short)0, uacc[1][t], false, false);
    }
  }

  // u0 = z^T (f32 state); v0 = tanh(z^T) carried directly as bf16 B-fragments
  v8f   ust[NT][NTILE];
  v16bf vfr[NT][NKC];
#pragma unroll
  for (int n = 0; n < NT; ++n) {
#pragma unroll
    for (int t = 0; t < NTILE; ++t) ust[n][t] = uacc[n][t];
#pragma unroll
    for (int kc = 0; kc < NKC; ++kc) {
      v16bf f;
#pragma unroll
      for (int e = 0; e < 8; ++e) {
        f[e]     = (bf16_t)fast_tanh(uacc[n][2 * kc][e]);
        f[e + 8] = (bf16_t)fast_tanh(uacc[n][2 * kc + 1][e]);
      }
      vfr[n][kc] = f;
    }
  }

  // =======================================================================
  // 4 blocks x 20 steps; each LDS W-fragment feeds 2 WMMAs (2 N-tiles)
  // =======================================================================
  for (int blk = 0; blk < NBLOCKS; ++blk) {
    const float* W  = (blk == 0) ? W1 : (blk == 1) ? W2 : (blk == 2) ? W3 : W4;
    const float* bv = (blk == 0) ? b1 : (blk == 1) ? b2 : (blk == 2) ? b3 : b4;

    __syncthreads();   // previous users of sWA/sWB done
    for (int idx = tid; idx < UNITS * UNITS; idx += 256) {
      int m = idx >> 7, k = idx & 127;
      sWB[idx] = (bf16_t)W[idx];              // [m][k] = W[m][k]  (phase B: W @ u^T)
      sWA[idx] = (bf16_t)W[k * UNITS + m];    // [m][k] = W[k][m]  (phase A: W^T @ v^T)
    }
    __syncthreads();

#pragma unroll
    for (int t = 0; t < NTILE; ++t) bias_v[t] = load_bias8(bv + t * 16 + 8 * hs);

    for (int s = 0; s < STEPS; ++s) {
      asm volatile("" ::: "memory");   // keep W-frag LDS loads inside the loop

      // -------- Phase A: du^T = W^T @ v^T + b ; u += DT*du --------
      v8f acc[NT][NTILE];
#pragma unroll
      for (int n = 0; n < NT; ++n)
#pragma unroll
        for (int t = 0; t < NTILE; ++t) acc[n][t] = bias_v[t];

#pragma unroll
      for (int kc = 0; kc < NKC; ++kc) {
        v16bf aw[NTILE];
#pragma unroll
        for (int t = 0; t < NTILE; ++t)
          aw[t] = lds_frag(sWA + (t * 16 + lr) * UNITS + kc * 32 + 8 * hs);
#pragma unroll
        for (int t = 0; t < NTILE; ++t) {
          acc[0][t] = __builtin_amdgcn_wmma_f32_16x16x32_bf16(
              false, aw[t], false, vfr[0][kc], (short)0, acc[0][t], false, false);
          acc[1][t] = __builtin_amdgcn_wmma_f32_16x16x32_bf16(
              false, aw[t], false, vfr[1][kc], (short)0, acc[1][t], false, false);
        }
      }
#pragma unroll
      for (int n = 0; n < NT; ++n)
#pragma unroll
        for (int t = 0; t < NTILE; ++t)
#pragma unroll
          for (int r = 0; r < 8; ++r) ust[n][t][r] += DT_C * acc[n][t][r];

      // -------- Phase B: dv^T = W @ u_new^T ; v = scale*(v - DT*dv + c*tanh(..)) --------
      v8f acc2[NT][NTILE];
#pragma unroll
      for (int n = 0; n < NT; ++n)
#pragma unroll
        for (int t = 0; t < NTILE; ++t) {
          v8f z;
#pragma unroll
          for (int r = 0; r < 8; ++r) z[r] = 0.0f;
          acc2[n][t] = z;
        }
#pragma unroll
      for (int kc = 0; kc < NKC; ++kc) {
        v16bf aw[NTILE];
#pragma unroll
        for (int t = 0; t < NTILE; ++t)
          aw[t] = lds_frag(sWB + (t * 16 + lr) * UNITS + kc * 32 + 8 * hs);
        v16bf uf0 = state_frag(ust[0][2 * kc], ust[0][2 * kc + 1]);   // transient
        v16bf uf1 = state_frag(ust[1][2 * kc], ust[1][2 * kc + 1]);
#pragma unroll
        for (int t = 0; t < NTILE; ++t) {
          acc2[0][t] = __builtin_amdgcn_wmma_f32_16x16x32_bf16(
              false, aw[t], false, uf0, (short)0, acc2[0][t], false, false);
          acc2[1][t] = __builtin_amdgcn_wmma_f32_16x16x32_bf16(
              false, aw[t], false, uf1, (short)0, acc2[1][t], false, false);
        }
      }
      // elementwise v update directly on the bf16 fragment carry
#pragma unroll
      for (int n = 0; n < NT; ++n)
#pragma unroll
        for (int kc = 0; kc < NKC; ++kc) {
          v16bf f = vfr[n][kc];
#pragma unroll
          for (int e = 0; e < 8; ++e) {
            float v0 = (float)f[e]     - DT_C * acc2[n][2 * kc][e];
            float v1 = (float)f[e + 8] - DT_C * acc2[n][2 * kc + 1][e];
            f[e]     = (bf16_t)(SCALE_C * (v0 + CC_C * fast_tanh(v0)));
            f[e + 8] = (bf16_t)(SCALE_C * (v1 + CC_C * fast_tanh(v1)));
          }
          vfr[n][kc] = f;
        }
    }
  }

  // =======================================================================
  // Head: logits = u @ Wo^T + bo ; softmax  (reuse first 128 KB as f32)
  // =======================================================================
  __syncthreads();   // all waves done reading sWA/sWB
#pragma unroll
  for (int n = 0; n < NT; ++n)
#pragma unroll
    for (int t = 0; t < NTILE; ++t) {
      float* orow = sOutF + (wave * 32 + n * 16 + lr) * UNITS + t * 16 + 8 * hs;
      *(float4*)(orow)     = make_float4(ust[n][t][0], ust[n][t][1], ust[n][t][2], ust[n][t][3]);
      *(float4*)(orow + 4) = make_float4(ust[n][t][4], ust[n][t][5], ust[n][t][6], ust[n][t][7]);
    }
  __syncthreads();

  {
    const int row = tid;                       // 256 threads, 256 rows
    const float* ur = sOutF + row * UNITS;
    float logits[OUTPUT_DIM];
    float mx = -INFINITY;
#pragma unroll
    for (int o = 0; o < OUTPUT_DIM; ++o) {
      float a = sBo[o];
      for (int k = 0; k < UNITS; ++k) a += ur[k] * sWo[o * UNITS + k];
      logits[o] = a;
      mx = fmaxf(mx, a);
    }
    float den = 0.0f;
#pragma unroll
    for (int o = 0; o < OUTPUT_DIM; ++o) { float e = __expf(logits[o] - mx); logits[o] = e; den += e; }
    float inv = 1.0f / den;
    float* op = out + (size_t)(blockIdx.x * TILE_ROWS + row) * OUTPUT_DIM;
#pragma unroll
    for (int o = 0; o < OUTPUT_DIM; ++o) op[o] = logits[o] * inv;
  }
}

extern "C" void kernel_launch(void* const* d_in, const int* in_sizes, int n_in,
                              void* d_out, int out_size, void* d_ws, size_t ws_size,
                              hipStream_t stream) {
  const float* x  = (const float*)d_in[0];
  const float* Wi = (const float*)d_in[1];
  const float* bi = (const float*)d_in[2];
  const float* W1 = (const float*)d_in[3];
  const float* b1 = (const float*)d_in[4];
  const float* W2 = (const float*)d_in[5];
  const float* b2 = (const float*)d_in[6];
  const float* W3 = (const float*)d_in[7];
  const float* b3 = (const float*)d_in[8];
  const float* W4 = (const float*)d_in[9];
  const float* b4 = (const float*)d_in[10];
  const float* Wo = (const float*)d_in[11];
  const float* bo = (const float*)d_in[12];
  float* out = (float*)d_out;

  const int batch = in_sizes[0] / INPUT_DIM;
  dim3 grid(batch / TILE_ROWS), block(256);
  hipLaunchKernelGGL(transnet_kernel, grid, block, 0, stream,
                     x, Wi, bi, W1, b1, W2, b2, W3, b3, W4, b4, Wo, bo, out);
}
